// swin_bra_54030688583942
// MI455X (gfx1250) — matrix operand order repaired and checked by reading, not compiled, not run on hardware
//
#include <hip/hip_runtime.h>

#define Bz 4
#define Hz 128
#define Wz 128
#define DIMz 128
#define NHz 8
#define TOPKz 4
#define C2z 64
#define NREGz 256
#define Lz 16384
#define HDz 16
#define SCALEz 0.08838834764831845f  // 128^-0.5

typedef __attribute__((ext_vector_type(16))) _Float16 v16h;
typedef __attribute__((ext_vector_type(8)))  float    v8f;

#if defined(__has_builtin)
#if __has_builtin(__builtin_amdgcn_tensor_load_to_lds)
#define HAVE_TDM 1
#endif
#endif
#ifndef HAVE_TDM
#define HAVE_TDM 0
#endif

__device__ __forceinline__ v8f wmma16(v16h a, v16h b, v8f c) {
  return __builtin_amdgcn_wmma_f32_16x16x32_f16(false, a, false, b, (short)0, c, false, false);
}

#if HAVE_TDM
typedef __attribute__((ext_vector_type(4))) unsigned int tdm_v4u;
typedef __attribute__((ext_vector_type(8))) int          tdm_v8i;
typedef __attribute__((ext_vector_type(4))) int          tdm_v4i;

// Issue one TDM 2D tile load (f32 elements): rows x row_elems contiguous tile of a
// row-major tensor whose row pitch == row_elems. D# per CDNA5 ISA 8.3/8.4.
__device__ __forceinline__ void tdm_load_2d_f32(unsigned lds_byte_off, const float* gsrc,
                                                unsigned rows, unsigned row_elems,
                                                unsigned tensor_rows) {
  const unsigned long long ga = (unsigned long long)(size_t)gsrc;
  tdm_v4u g0;
  g0[0] = 1u;                                           // count=1 (valid user D#)
  g0[1] = lds_byte_off;                                 // lds_addr
  g0[2] = (unsigned)(ga & 0xffffffffu);                 // global_addr[31:0]
  g0[3] = (unsigned)((ga >> 32) & 0x1ffffffu) | (2u << 30); // global_addr[56:32] | type=2
  tdm_v8i g1;
  g1[0] = (int)(2u << 16);                              // workgroup_mask=0, data_size=4B
  g1[1] = (int)((row_elems & 0xffffu) << 16);           // tensor_dim0[15:0]
  g1[2] = (int)(((row_elems >> 16) & 0xffffu) | ((tensor_rows & 0xffffu) << 16)); // dim0 hi | dim1 lo
  g1[3] = (int)(((tensor_rows >> 16) & 0xffffu) | ((row_elems & 0xffffu) << 16)); // dim1 hi | tile_dim0
  g1[4] = (int)(rows & 0xffffu);                        // tile_dim1 | tile_dim2=0
  g1[5] = (int)row_elems;                               // tensor_dim0_stride[31:0]
  g1[6] = 0;                                            // stride0 hi | dim1_stride lo
  g1[7] = 0;                                            // dim1_stride hi
  tdm_v4i z4 = {0, 0, 0, 0};
#if __clang_major__ >= 23
  tdm_v8i z8 = {0, 0, 0, 0, 0, 0, 0, 0};
  __builtin_amdgcn_tensor_load_to_lds(g0, g1, z4, z4, z8, 0);
#else
  __builtin_amdgcn_tensor_load_to_lds(g0, g1, z4, z4, 0);
#endif
}
#endif

// ---------- WMMA fragment loaders (wave32, contiguous k order k = 16*(lane/16) + e) ----------
// A 16x32 (MxK): lane<16 -> M=lane, K elems 0..15; lane>=16 -> M=lane-16, K 16..31
__device__ __forceinline__ v16h afrag_h(const _Float16* s, int ld, int lane) {
  const int m = lane & 15, kh = (lane & 16) ? 16 : 0;
  v16h a;
#pragma unroll
  for (int e = 0; e < 16; ++e) {
    const int k = kh + e;
    a[e] = s[m * ld + k];
  }
  return a;
}
__device__ __forceinline__ v16h afrag_h_scaled(const _Float16* s, int ld, int lane, int kmax, float scl) {
  const int m = lane & 15, kh = (lane & 16) ? 16 : 0;
  v16h a;
#pragma unroll
  for (int e = 0; e < 16; ++e) {
    const int k = kh + e;
    a[e] = (k < kmax) ? (_Float16)((float)s[m * ld + k] * scl) : (_Float16)0.f;
  }
  return a;
}
__device__ __forceinline__ v16h afrag_f(const float* s, int ld, int lane) {
  const int m = lane & 15, kh = (lane & 16) ? 16 : 0;
  v16h a;
#pragma unroll
  for (int e = 0; e < 16; ++e) {
    const int k = kh + e;
    a[e] = (_Float16)s[m * ld + k];
  }
  return a;
}
// B 32x16 (KxN) from row-major KxN source: lane<16 -> N=lane,K=0..15; lane>=16 -> N=lane-16,K=16..31
__device__ __forceinline__ v16h bfrag_f(const float* s, int ld, int lane) {
  const int n = lane & 15, kh = (lane & 16) ? 16 : 0;
  v16h b;
#pragma unroll
  for (int e = 0; e < 16; ++e) b[e] = (_Float16)s[(kh + e) * ld + n];
  return b;
}
__device__ __forceinline__ v16h bfrag_h_nmax(const _Float16* s, int ld, int lane, int nmax) {
  const int n = lane & 15, kh = (lane & 16) ? 16 : 0;
  v16h b;
#pragma unroll
  for (int e = 0; e < 16; ++e) b[e] = (n < nmax) ? s[(kh + e) * ld + n] : (_Float16)0.f;
  return b;
}
// B from transposed (NxK row-major) f16 source, zero-padded past kmax
__device__ __forceinline__ v16h bTfrag_h(const _Float16* s, int ld, int lane, int kmax) {
  const int n = lane & 15, kh = (lane & 16) ? 16 : 0;
  v16h b;
#pragma unroll
  for (int e = 0; e < 16; ++e) {
    const int k = kh + e;
    b[e] = (k < kmax) ? s[n * ld + k] : (_Float16)0.f;
  }
  return b;
}

// ---------------- Kernel 1: fused window attention + both QKV GEMMs ----------------
__global__ __launch_bounds__(256) void k_winattn(
    const float* __restrict__ x,
    const float* __restrict__ wqkv_w, const float* __restrict__ wqkv_b,
    const float* __restrict__ bqkv_w, const float* __restrict__ bqkv_b,
    float* __restrict__ woutg, float* __restrict__ lcev1,
    float* __restrict__ qb, float* __restrict__ kb, float* __restrict__ vb)
{
  __shared__ _Float16 xs[64][64], xs2[64][64];
  __shared__ _Float16 qs[64][64], ks[64][64], vs[64][64];
  __shared__ float sc[64][64];

  const int blk = blockIdx.x;
  const int b = blk >> 8, reg = blk & 255;
  const int hr = reg >> 4, wr = reg & 15;
  const int tid = threadIdx.x, lane = tid & 31, wave = tid >> 5;

  // pull both qkv weight matrices toward GL2 early (reused by all 1024 blocks)
  __builtin_prefetch(wqkv_w + tid * 48, 0, 1);
  __builtin_prefetch(bqkv_w + tid * 48, 0, 1);

  // load 64 tokens x 128 channels of x; split halves to f16 LDS
  for (int i = tid; i < 64 * 32; i += 256) {
    const int t = i >> 5, c4 = (i & 31) << 2;
    const int y = hr * 8 + (t >> 3), xx = wr * 8 + (t & 7);
    const float4 f = *(const float4*)(x + ((((size_t)b * Hz + y) * Wz + xx) << 7) + c4);
    _Float16* d = (c4 < 64) ? &xs[t][c4] : &xs2[t][c4 - 64];
    d[0] = (_Float16)f.x; d[1] = (_Float16)f.y; d[2] = (_Float16)f.z; d[3] = (_Float16)f.w;
  }
  __syncthreads();

  // window qkv: (64x64)@(64x192)+b -> qs|ks|vs (f16, LDS)
  for (int tt = wave; tt < 48; tt += 8) {
    const int mt = tt / 12, nt = tt % 12;
    v8f c = {};
#pragma unroll
    for (int k0 = 0; k0 < 64; k0 += 32) {
      v16h a  = afrag_h(&xs[mt * 16][k0], 64, lane);
      v16h bb = bfrag_f(wqkv_w + (size_t)k0 * 192 + nt * 16, 192, lane);
      c = wmma16(a, bb, c);
    }
    const int nn = lane & 15, mh = (lane >> 4) * 8;
    const int j = nt * 16 + nn;
    const float bj = wqkv_b[j];
    const int which = j >> 6, cc = j & 63;
    _Float16 (*dst)[64] = (which == 0) ? qs : (which == 1) ? ks : vs;
#pragma unroll
    for (int r = 0; r < 8; ++r) dst[mt * 16 + mh + r][cc] = (_Float16)(c[r] + bj);
  }
  // BRA qkv: (64x64)@(64x192)+b -> global q/k/v (f32, image layout [b][y][x][c])
  for (int tt = wave; tt < 48; tt += 8) {
    const int mt = tt / 12, nt = tt % 12;
    v8f c = {};
#pragma unroll
    for (int k0 = 0; k0 < 64; k0 += 32) {
      v16h a  = afrag_h(&xs2[mt * 16][k0], 64, lane);
      v16h bb = bfrag_f(bqkv_w + (size_t)k0 * 192 + nt * 16, 192, lane);
      c = wmma16(a, bb, c);
    }
    const int nn = lane & 15, mh = (lane >> 4) * 8;
    const int j = nt * 16 + nn;
    const float bj = bqkv_b[j];
    const int which = j >> 6, cc = j & 63;
    float* dst = (which == 0) ? qb : (which == 1) ? kb : vb;
#pragma unroll
    for (int r = 0; r < 8; ++r) {
      const int t = mt * 16 + mh + r;
      const int y = hr * 8 + (t >> 3), xx = wr * 8 + (t & 7);
      dst[(((size_t)b * Hz + y) * Wz + xx) * C2z + cc] = c[r] + bj;
    }
  }
  __syncthreads();

  // lcev1 = window v back to image layout
  for (int i = tid; i < 4096; i += 256) {
    const int t = i >> 6, cc = i & 63;
    const int y = hr * 8 + (t >> 3), xx = wr * 8 + (t & 7);
    lcev1[(((size_t)b * Hz + y) * Wz + xx) * C2z + cc] = (float)vs[t][cc];
  }

  // per-head attention (head dim 8, zero-padded into K=32 / N=16 tiles)
  for (int h = 0; h < 8; ++h) {
    for (int tt = wave; tt < 16; tt += 8) {
      const int mt = tt >> 2, nt = tt & 3;
      v16h a  = afrag_h_scaled(&qs[mt * 16][h * 8], 64, lane, 8, SCALEz);
      v16h bb = bTfrag_h(&ks[nt * 16][h * 8], 64, lane, 8);
      v8f c = {};
      c = wmma16(a, bb, c);
      const int nn = lane & 15, mh = (lane >> 4) * 8;
#pragma unroll
      for (int r = 0; r < 8; ++r) sc[mt * 16 + mh + r][nt * 16 + nn] = c[r];
    }
    __syncthreads();
    if (tid < 64) {
      float mx = -3.4e38f;
      for (int j = 0; j < 64; ++j) mx = fmaxf(mx, sc[tid][j]);
      float s = 0.f;
      for (int j = 0; j < 64; ++j) { const float e = __expf(sc[tid][j] - mx); sc[tid][j] = e; s += e; }
      const float inv = 1.f / s;
      for (int j = 0; j < 64; ++j) sc[tid][j] *= inv;
    }
    __syncthreads();
    if (wave < 4) {
      const int mt = wave;
      v8f c = {};
#pragma unroll
      for (int k0 = 0; k0 < 64; k0 += 32) {
        v16h a  = afrag_f(&sc[mt * 16][k0], 64, lane);
        v16h bb = bfrag_h_nmax(&vs[k0][h * 8], 64, lane, 8);
        c = wmma16(a, bb, c);
      }
      const int nn = lane & 15, mh = (lane >> 4) * 8;
      if (nn < 8) {
#pragma unroll
        for (int r = 0; r < 8; ++r) {
          const int t = mt * 16 + mh + r;
          const int y = hr * 8 + (t >> 3), xx = wr * 8 + (t & 7);
          woutg[(((size_t)b * Hz + y) * Wz + xx) * C2z + h * 8 + nn] = c[r];
        }
      }
    }
    __syncthreads();
  }
}

// ---------------- Kernel 2: region pooling ----------------
__global__ __launch_bounds__(64) void k_pool(const float* __restrict__ qb, const float* __restrict__ kb,
                                             float* __restrict__ qr, float* __restrict__ kr) {
  const int b = blockIdx.x >> 8, reg = blockIdx.x & 255;
  const int hr = reg >> 4, wr = reg & 15, c = threadIdx.x;
  float sq = 0.f, sk = 0.f;
  for (int t = 0; t < 64; ++t) {
    const int y = hr * 8 + (t >> 3), xx = wr * 8 + (t & 7);
    const size_t i = (((size_t)b * Hz + y) * Wz + xx) * C2z + c;
    sq += qb[i]; sk += kb[i];
  }
  qr[((size_t)b * NREGz + reg) * C2z + c] = sq * (1.f / 64.f);
  kr[((size_t)b * NREGz + reg) * C2z + c] = sk * (1.f / 64.f);
}

// ---------------- Kernel 3: region affinity + top-4 ----------------
__global__ __launch_bounds__(256) void k_topk(const float* __restrict__ qr, const float* __restrict__ kr,
                                              int* __restrict__ idxg) {
  __shared__ float av[256];
  const int b = blockIdx.x >> 8, i = blockIdx.x & 255, j = threadIdx.x;
  const float* q = qr + ((size_t)b * NREGz + i) * C2z;
  const float* k = kr + ((size_t)b * NREGz + j) * C2z;
  float s = 0.f;
  for (int c = 0; c < C2z; ++c) s += q[c] * k[c];
  av[j] = s;
  __syncthreads();
  if (j == 0) {
    for (int t = 0; t < TOPKz; ++t) {
      float best = -3.4e38f; int bi = 0;
      for (int r = 0; r < 256; ++r) if (av[r] > best) { best = av[r]; bi = r; }
      idxg[((b << 8) + i) * TOPKz + t] = bi;
      av[bi] = -3.4e38f;
    }
  }
}

// ---------------- Kernel 4: region (BRA) attention ----------------
__global__ __launch_bounds__(256) void k_regattn(
    const float* __restrict__ qb, const float* __restrict__ kb, const float* __restrict__ vb,
    const int* __restrict__ idxg, float* __restrict__ brao)
{
  __shared__ float sc[64][256];
  __shared__ _Float16 q5[64][8], kg[256][8], vg[256][8];
  __shared__ int sidx[4];
  const int blk = blockIdx.x;
  const int r = blk & 255, bn = blk >> 8, hN = bn & 7, b = bn >> 3;
  const int hr = r >> 4, wr = r & 15;
  const int tid = threadIdx.x, lane = tid & 31, wave = tid >> 5;
  if (tid < 4) sidx[tid] = idxg[((b << 8) + r) * TOPKz + tid];
  for (int i = tid; i < 512; i += 256) {
    const int t = i >> 3, dd = i & 7;
    const int y = hr * 8 + (t >> 3), xx = wr * 8 + (t & 7);
    q5[t][dd] = (_Float16)qb[(((size_t)b * Hz + y) * Wz + xx) * C2z + hN * 8 + dd];
  }
  __syncthreads();
  for (int i = tid; i < 2048; i += 256) {
    const int kk = i >> 3, dd = i & 7;
    const int rg = sidx[kk >> 6], t = kk & 63;
    const int y = (rg >> 4) * 8 + (t >> 3), xx = (rg & 15) * 8 + (t & 7);
    const size_t gi = (((size_t)b * Hz + y) * Wz + xx) * C2z + hN * 8 + dd;
    kg[kk][dd] = (_Float16)kb[gi];
    vg[kk][dd] = (_Float16)vb[gi];
  }
  __syncthreads();
  for (int tt = wave; tt < 64; tt += 8) {
    const int mt = tt >> 4, nt = tt & 15;
    v16h a  = afrag_h_scaled(&q5[mt * 16][0], 8, lane, 8, SCALEz);
    v16h bb = bTfrag_h(&kg[nt * 16][0], 8, lane, 8);
    v8f c = {};
    c = wmma16(a, bb, c);
    const int nn = lane & 15, mh = (lane >> 4) * 8;
#pragma unroll
    for (int rr = 0; rr < 8; ++rr) sc[mt * 16 + mh + rr][nt * 16 + nn] = c[rr];
  }
  __syncthreads();
  if (tid < 64) {
    float mx = -3.4e38f;
    for (int j = 0; j < 256; ++j) mx = fmaxf(mx, sc[tid][j]);
    float s = 0.f;
    for (int j = 0; j < 256; ++j) { const float e = __expf(sc[tid][j] - mx); sc[tid][j] = e; s += e; }
    const float inv = 1.f / s;
    for (int j = 0; j < 256; ++j) sc[tid][j] *= inv;
  }
  __syncthreads();
  if (wave < 4) {
    const int mt = wave;
    v8f c = {};
    for (int k0 = 0; k0 < 256; k0 += 32) {
      v16h a  = afrag_f(&sc[mt * 16][k0], 256, lane);
      v16h bb = bfrag_h_nmax(&vg[k0][0], 8, lane, 8);
      c = wmma16(a, bb, c);
    }
    const int nn = lane & 15, mh = (lane >> 4) * 8;
    if (nn < 8) {
#pragma unroll
      for (int rr = 0; rr < 8; ++rr) {
        const int t = mt * 16 + mh + rr;
        const int y = hr * 8 + (t >> 3), xx = wr * 8 + (t & 7);
        brao[(((size_t)b * Hz + y) * Wz + xx) * C2z + hN * 8 + nn] = c[rr];
      }
    }
  }
}

// ---------------- Kernel 5: depthwise LePE + combine -> out (B,L,128) ----------------
__global__ __launch_bounds__(256) void k_lepe(
    const float* __restrict__ woutg, const float* __restrict__ brao,
    const float* __restrict__ vb, const float* __restrict__ lcev1,
    const float* __restrict__ lw, const float* __restrict__ lbias,
    float* __restrict__ xout)
{
  const size_t g = (size_t)blockIdx.x * 256 + threadIdx.x;
  const int c = (int)(g & 127);
  const size_t p = g >> 7;
  const int xx = (int)(p & 127), y = (int)((p >> 7) & 127), b = (int)(p >> 14);
  const float base = (c < 64) ? woutg[p * 64 + c] : brao[p * 64 + (c - 64)];
  const float* wc = lw + c * 9;
  const float* src = (c < 64) ? vb : lcev1;
  const int cc = (c < 64) ? c : c - 64;
  float acc = lbias[c];
#pragma unroll
  for (int dy = -1; dy <= 1; ++dy) {
    const int yy = y + dy;
    if (yy < 0 || yy >= Hz) continue;
#pragma unroll
    for (int dx = -1; dx <= 1; ++dx) {
      const int xw = xx + dx;
      if (xw < 0 || xw >= Wz) continue;
      acc += wc[(dy + 1) * 3 + (dx + 1)] * src[(((size_t)b * Hz + yy) * Wz + xw) * C2z + cc];
    }
  }
  xout[g] = base + acc;
}

// ---------------- Kernel 6: ca_qkv GEMM (65536x128 @ 128x384), TDM-staged A ----------------
__global__ __launch_bounds__(256) void k_caqkv(
    const float* __restrict__ xin, const float* __restrict__ w, const float* __restrict__ bias,
    float* __restrict__ cq, float* __restrict__ ck, float* __restrict__ cv)
{
  __shared__ float as[64][128];
  const int row0 = blockIdx.x * 64;
  const int tid = threadIdx.x, lane = tid & 31, wave = tid >> 5;
  __builtin_prefetch(w + tid * 192, 0, 1);   // 128x384 weights -> GL2
#if HAVE_TDM
  if (wave == 0) {
    tdm_load_2d_f32((unsigned)(size_t)&as[0][0], xin + ((size_t)row0 << 7), 64, 128, (unsigned)Bz * Lz);
    __builtin_amdgcn_s_wait_tensorcnt(0);
  }
#else
  for (int i = tid; i < 64 * 32; i += 256) {
    const int rr = i >> 5, c4 = (i & 31) << 2;
    *(float4*)&as[rr][c4] = *(const float4*)(xin + ((size_t)(row0 + rr) << 7) + c4);
  }
#endif
  __syncthreads();
  for (int tt = wave; tt < 96; tt += 8) {
    const int mt = tt / 24, nt = tt % 24;
    v8f c = {};
#pragma unroll
    for (int k0 = 0; k0 < 128; k0 += 32) {
      v16h a  = afrag_f(&as[mt * 16][k0], 128, lane);
      v16h bb = bfrag_f(w + (size_t)k0 * 384 + nt * 16, 384, lane);
      c = wmma16(a, bb, c);
    }
    const int nn = lane & 15, mh = (lane >> 4) * 8;
    const int j = nt * 16 + nn;
    const float bj = bias[j];
    const int which = j >> 7, rest = j & 127, h = rest >> 4, dd = rest & 15;
    float* dst = (which == 0) ? cq : (which == 1) ? ck : cv;
#pragma unroll
    for (int r = 0; r < 8; ++r) {
      const int row = row0 + mt * 16 + mh + r;
      const int b = row >> 14, l = row & 16383;
      dst[(((size_t)(b * NHz + h)) * HDz + dd) * Lz + l] = c[r] + bj;
    }
  }
}

// ---------------- Kernel 7: XCA (channel attention) per (b,head) ----------------
__global__ __launch_bounds__(256) void k_xca(
    const float* __restrict__ cq, const float* __restrict__ ck, const float* __restrict__ cv,
    const float* __restrict__ temp, float* __restrict__ coutb)
{
  __shared__ float qt[16][128], kt[16][128], vt[16][128];
  __shared__ float gram[16][16], attn[16][16], nrm[32];
  const int b = blockIdx.x >> 3, h = blockIdx.x & 7;
  const int tid = threadIdx.x;
  const size_t base = ((size_t)(b * NHz + h) * HDz) * Lz;
  const float* qp = cq + base;
  const float* kp = ck + base;
  const float* vp = cv + base;
  const int dd = tid & 15, ee = tid >> 4;
  float g = 0.f, nacc = 0.f;
  for (int l0 = 0; l0 < Lz; l0 += 128) {
    for (int i = tid; i < 16 * 32; i += 256) {
      const int rr = i >> 5, c4 = (i & 31) << 2;
      *(float4*)&qt[rr][c4] = *(const float4*)(qp + (size_t)rr * Lz + l0 + c4);
      *(float4*)&kt[rr][c4] = *(const float4*)(kp + (size_t)rr * Lz + l0 + c4);
    }
    __syncthreads();
    for (int l = 0; l < 128; ++l) g += qt[dd][l] * kt[ee][l];
    if (tid < 16) { float s = 0.f; for (int l = 0; l < 128; ++l) s += qt[tid][l] * qt[tid][l]; nacc += s; }
    else if (tid < 32) { const int r = tid - 16; float s = 0.f; for (int l = 0; l < 128; ++l) s += kt[r][l] * kt[r][l]; nacc += s; }
    __syncthreads();
  }
  if (tid < 32) nrm[tid] = fmaxf(sqrtf(nacc), 1e-12f);
  __syncthreads();
  gram[dd][ee] = g / (nrm[dd] * nrm[16 + ee]) * temp[h];
  __syncthreads();
  if (tid < 16) {
    float mx = -3.4e38f;
    for (int j = 0; j < 16; ++j) mx = fmaxf(mx, gram[tid][j]);
    float s = 0.f;
    for (int j = 0; j < 16; ++j) { const float ex = __expf(gram[tid][j] - mx); attn[tid][j] = ex; s += ex; }
    const float inv = 1.f / s;
    for (int j = 0; j < 16; ++j) attn[tid][j] *= inv;
  }
  __syncthreads();
  const int d2 = tid >> 4, lb2 = (tid & 15) << 3;
  for (int l0 = 0; l0 < Lz; l0 += 128) {
    for (int i = tid; i < 512; i += 256) {
      const int rr = i >> 5, c4 = (i & 31) << 2;
      *(float4*)&vt[rr][c4] = *(const float4*)(vp + (size_t)rr * Lz + l0 + c4);
    }
    __syncthreads();
#pragma unroll
    for (int u = 0; u < 8; ++u) {
      float acc = 0.f;
#pragma unroll
      for (int j = 0; j < 16; ++j) acc += attn[d2][j] * vt[j][lb2 + u];
      coutb[((size_t)b * Lz + (l0 + lb2 + u)) * DIMz + h * HDz + d2] = acc;
    }
    __syncthreads();
  }
}

// ---------------- Kernel 8: output projection (65536x128 @ 128x128), TDM-staged A ----------------
__global__ __launch_bounds__(256) void k_proj(
    const float* __restrict__ xin, const float* __restrict__ w, const float* __restrict__ bias,
    float* __restrict__ out)
{
  __shared__ float as[64][128];
  const int row0 = blockIdx.x * 64;
  const int tid = threadIdx.x, lane = tid & 31, wave = tid >> 5;
  __builtin_prefetch(w + tid * 64, 0, 1);    // 128x128 weights -> GL2
#if HAVE_TDM
  if (wave == 0) {
    tdm_load_2d_f32((unsigned)(size_t)&as[0][0], xin + ((size_t)row0 << 7), 64, 128, (unsigned)Bz * Lz);
    __builtin_amdgcn_s_wait_tensorcnt(0);
  }
#else
  for (int i = tid; i < 64 * 32; i += 256) {
    const int rr = i >> 5, c4 = (i & 31) << 2;
    *(float4*)&as[rr][c4] = *(const float4*)(xin + ((size_t)(row0 + rr) << 7) + c4);
  }
#endif
  __syncthreads();
  for (int tt = wave; tt < 32; tt += 8) {
    const int mt = tt >> 3, nt = tt & 7;
    v8f c = {};
#pragma unroll
    for (int k0 = 0; k0 < 128; k0 += 32) {
      v16h a  = afrag_f(&as[mt * 16][k0], 128, lane);
      v16h bb = bfrag_f(w + (size_t)k0 * 128 + nt * 16, 128, lane);
      c = wmma16(a, bb, c);
    }
    const int nn = lane & 15, mh = (lane >> 4) * 8;
    const int j = nt * 16 + nn;
    const float bj = bias[j];
#pragma unroll
    for (int r = 0; r < 8; ++r)
      out[((size_t)(row0 + mt * 16 + mh + r) << 7) + j] = c[r] + bj;
  }
}

extern "C" void kernel_launch(void* const* d_in, const int* in_sizes, int n_in,
                              void* d_out, int out_size, void* d_ws, size_t ws_size,
                              hipStream_t stream) {
  (void)in_sizes; (void)n_in; (void)out_size; (void)ws_size;
  const float* x         = (const float*)d_in[0];
  const float* wqkv_w    = (const float*)d_in[1];
  const float* wqkv_b    = (const float*)d_in[2];
  const float* bqkv_w    = (const float*)d_in[3];
  const float* bqkv_b    = (const float*)d_in[4];
  const float* lepe_w    = (const float*)d_in[5];
  const float* lepe_b    = (const float*)d_in[6];
  const float* ca_qkv_w  = (const float*)d_in[7];
  const float* ca_qkv_b  = (const float*)d_in[8];
  const float* ca_temp   = (const float*)d_in[9];
  const float* ca_proj_w = (const float*)d_in[10];
  const float* ca_proj_b = (const float*)d_in[11];

  float* ws = (float*)d_ws;
  const size_t P64 = (size_t)Bz * Hz * Wz * 64;            // 4,194,304 floats
  float* woutg = ws + 0 * P64;
  float* lcev1 = ws + 1 * P64;
  float* qb    = ws + 2 * P64;
  float* kb    = ws + 3 * P64;
  float* vb    = ws + 4 * P64;
  float* brao  = ws + 5 * P64;
  float* qr    = ws + 6 * P64;                             // B*256*64
  float* kr    = qr + (size_t)Bz * NREGz * C2z;
  int*   idx   = (int*)(kr + (size_t)Bz * NREGz * C2z);
  float* xout  = ws + 6 * P64 + 262144;                    // B*L*128
  float* cq    = ws + 0;                                   // reuses phase-A region (dead by then)
  float* ck    = cq + (size_t)Bz * NHz * HDz * Lz;
  float* cv    = ck + (size_t)Bz * NHz * HDz * Lz;
  float* coutb = xout + (size_t)Bz * Lz * DIMz;

  k_winattn<<<1024, 256, 0, stream>>>(x, wqkv_w, wqkv_b, bqkv_w, bqkv_b,
                                      woutg, lcev1, qb, kb, vb);
  k_pool<<<1024, 64, 0, stream>>>(qb, kb, qr, kr);
  k_topk<<<1024, 256, 0, stream>>>(qr, kr, idx);
  k_regattn<<<8192, 256, 0, stream>>>(qb, kb, vb, idx, brao);
  k_lepe<<<32768, 256, 0, stream>>>(woutg, brao, vb, lcev1, lepe_w, lepe_b, xout);
  k_caqkv<<<1024, 256, 0, stream>>>(xout, ca_qkv_w, ca_qkv_b, cq, ck, cv);
  k_xca<<<32, 256, 0, stream>>>(cq, ck, cv, ca_temp, coutb);
  k_proj<<<1024, 256, 0, stream>>>(coutb, ca_proj_w, ca_proj_b, (float*)d_out);
}
